// FTModule_37830071943323
// MI455X (gfx1250) — matrix operand order repaired and checked by reading, not compiled, not run on hardware
//
#include <hip/hip_runtime.h>
#include <hip/hip_bf16.h>

typedef __attribute__((ext_vector_type(16))) __bf16 v16bf;
typedef __attribute__((ext_vector_type(8)))  __bf16 v8bf;
typedef __attribute__((ext_vector_type(8)))  float  v8f;
typedef __attribute__((ext_vector_type(4)))  float  v4f;

union BF16Frag { v16bf v; v8bf h[2]; };

__device__ __forceinline__ unsigned short f2bf(float f) {
    unsigned int u = __float_as_uint(f);
    unsigned int r = u + 0x7FFFu + ((u >> 16) & 1u);   // round-to-nearest-even
    return (unsigned short)(r >> 16);
}

// ---------------------------------------------------------------------------
// Kernel 0: convert the four GRU weight matrices to bf16 in workspace.
// Layout (ushort offsets): WihT[0,24576) WhhT[24576,73728) WihF[73728,98304)
// WhhF[98304,147456)
// ---------------------------------------------------------------------------
__global__ __launch_bounds__(256)
void prep_weights(const float* __restrict__ wih_t, const float* __restrict__ whh_t,
                  const float* __restrict__ wih_f, const float* __restrict__ whh_f,
                  unsigned short* __restrict__ out)
{
    int i = blockIdx.x * 256 + threadIdx.x;
    if (i < 24576)        out[i] = f2bf(wih_t[i]);
    else if (i < 73728)   out[i] = f2bf(whh_t[i - 24576]);
    else if (i < 98304)   out[i] = f2bf(wih_f[i - 73728]);
    else if (i < 147456)  out[i] = f2bf(whh_f[i - 98304]);
}

// ---------------------------------------------------------------------------
// Kernel 1: both GRUs. blockIdx 0..127 = time GRU tiles, 128..255 = freq GRU.
// Each workgroup (8 wave32) carries 16 sequences through all 256 steps.
// Per step: gi = x_t * Wih^T (K=64), gh = h * Whh^T (K=128) via bf16 WMMA,
// weight B-fragments live in registers across the whole recurrence.
// ---------------------------------------------------------------------------
__global__ __launch_bounds__(256)
void gru_kernel(const float* __restrict__ x,
                const unsigned short* __restrict__ WihT, const unsigned short* __restrict__ WhhT,
                const float* __restrict__ bihT, const float* __restrict__ bhhT,
                const unsigned short* __restrict__ WihF, const unsigned short* __restrict__ WhhF,
                const float* __restrict__ bihF, const float* __restrict__ bhhF,
                float* __restrict__ hTt, float* __restrict__ hTf)
{
    __shared__ __attribute__((aligned(16))) unsigned short xbuf[16][64];   // bf16 x tile
    __shared__ __attribute__((aligned(16))) unsigned short hbuf[16][128];  // bf16 h
    __shared__ float hflt[16][128];          // f32 h (master copy)
    __shared__ float gsum[16][268];          // gi+gh for r,z gates (g 0..255)
    __shared__ float gin [16][140];          // i_n  (g 256..383)
    __shared__ float ghn [16][140];          // W_hn h + (bias added later)
    __shared__ float bsr[128], bsz[128], bnn[128], bhn[128];

    const int tid  = threadIdx.x;
    const int gru  = blockIdx.x >> 7;        // 0 = time, 1 = freq
    const int tile = blockIdx.x & 127;

    const unsigned short* Wih = gru ? WihF : WihT;
    const unsigned short* Whh = gru ? WhhF : WhhT;
    const float* bih = gru ? bihF : bihT;
    const float* bhh = gru ? bhhF : bhhT;
    float* hout = (gru ? hTf : hTt) + (size_t)(tile * 16) * 128;

    const int b  = tile >> 4;                // batch
    const int p0 = (tile & 15) << 4;         // f0 (time GRU) or t0 (freq GRU)

    // ---- init biases + h = 0 ----
    if (tid < 128) {
        bsr[tid] = bih[tid]       + bhh[tid];
        bsz[tid] = bih[128 + tid] + bhh[128 + tid];
        bnn[tid] = bih[256 + tid];
        bhn[tid] = bhh[256 + tid];
    }
    for (int e = tid; e < 2048; e += 256) {
        (&hflt[0][0])[e] = 0.0f;
        (&hbuf[0][0])[e] = 0;
    }

    const int lane = tid & 31;
    const int wid  = tid >> 5;
    const int nn_  = lane & 15;              // B/C/D column within tile
    const int hi   = (lane >> 4) & 1;
    const int aoff = hi ? 16 : 0;            // A-fragment K base (lanes 16-31)
    const int boff = hi ? 16 : 0;            // B-fragment K base (lanes 16-31)
    const int rowoff = hi ? 8 : 0;           // C/D rows 8..15 on high lanes

    // ---- preload weight B-fragments into registers (held across all steps) ----
    v16bf wih_fr[3][2];
    v16bf whh_fr[3][4];
#pragma unroll
    for (int jj = 0; jj < 3; ++jj) {
        const int g0 = (wid * 3 + jj) << 4;
#pragma unroll
        for (int kt = 0; kt < 2; ++kt)
            wih_fr[jj][kt] = *(const v16bf*)(Wih + (g0 + nn_) * 64  + kt * 32 + boff);
#pragma unroll
        for (int kt = 0; kt < 4; ++kt)
            whh_fr[jj][kt] = *(const v16bf*)(Whh + (g0 + nn_) * 128 + kt * 32 + boff);
    }

    __syncthreads();

    const int lc = tid >> 2;                 // input channel 0..63
    const int lm = (tid & 3) << 2;           // sequence-in-tile quad base

    for (int step = 0; step < 256; ++step) {
        // ---- 1. stage x tile [16 seq][64 ch] as bf16 ----
        if (gru == 0) {
            // t_in(s=b*256+f, t, c) = x[b,c,t,f]; f contiguous -> float4
            const float* p = x + (((size_t)(b * 64 + lc) * 256 + step) * 256 + p0 + lm);
            v4f v = *(const v4f*)p;
            xbuf[lm + 0][lc] = f2bf(v[0]);
            xbuf[lm + 1][lc] = f2bf(v[1]);
            xbuf[lm + 2][lc] = f2bf(v[2]);
            xbuf[lm + 3][lc] = f2bf(v[3]);
            if (step + 1 < 256) __builtin_prefetch(p + 256, 0, 3);  // next timestep row
        } else {
            // f_in(s=b*256+t, f, c) = x[b,c,t,f]; strided gather (L2-resident)
            const float* p = x + (((size_t)(b * 64 + lc) * 256 + p0 + lm) * 256 + step);
            xbuf[lm + 0][lc] = f2bf(p[0]);
            xbuf[lm + 1][lc] = f2bf(p[256]);
            xbuf[lm + 2][lc] = f2bf(p[512]);
            xbuf[lm + 3][lc] = f2bf(p[768]);
        }
        __syncthreads();

        // ---- 2. WMMA: each wave owns 3 of the 24 gate tiles ----
        const int mr = nn_;                  // A-fragment row = lane & 15
#pragma unroll
        for (int jj = 0; jj < 3; ++jj) {
            const int j  = wid * 3 + jj;
            const int g0 = j << 4;
            v8f accI, accH;
#pragma unroll
            for (int q = 0; q < 8; ++q) { accI[q] = 0.0f; accH[q] = 0.0f; }

            // gi tile: [16x64] x [64x16]
#pragma unroll
            for (int kt = 0; kt < 2; ++kt) {
                BF16Frag a;
                a.h[0] = *(const v8bf*)&xbuf[mr][kt * 32 + aoff];
                a.h[1] = *(const v8bf*)&xbuf[mr][kt * 32 + aoff + 8];
                accI = __builtin_amdgcn_wmma_f32_16x16x32_bf16(
                    false, a.v, false, wih_fr[jj][kt], (short)0, accI, false, false);
            }
            // gh tile: [16x128] x [128x16]
#pragma unroll
            for (int kt = 0; kt < 4; ++kt) {
                BF16Frag a;
                a.h[0] = *(const v8bf*)&hbuf[mr][kt * 32 + aoff];
                a.h[1] = *(const v8bf*)&hbuf[mr][kt * 32 + aoff + 8];
                accH = __builtin_amdgcn_wmma_f32_16x16x32_bf16(
                    false, a.v, false, whh_fr[jj][kt], (short)0, accH, false, false);
            }

            if (j < 16) {       // r,z gates: only the sum matters
#pragma unroll
                for (int q = 0; q < 8; ++q)
                    gsum[rowoff + q][g0 + nn_] = accI[q] + accH[q];
            } else {            // n gate: keep input/hidden parts separate
                const int gc = g0 - 256;
#pragma unroll
                for (int q = 0; q < 8; ++q) {
                    gin[rowoff + q][gc + nn_] = accI[q];
                    ghn[rowoff + q][gc + nn_] = accH[q];
                }
            }
        }
        __syncthreads();

        // ---- 3. elementwise GRU update (f32 master h) ----
        for (int e = tid; e < 2048; e += 256) {
            const int m = e >> 7, h = e & 127;
            float r  = 1.0f / (1.0f + __expf(-(gsum[m][h]       + bsr[h])));
            float z  = 1.0f / (1.0f + __expf(-(gsum[m][128 + h] + bsz[h])));
            float nv = tanhf(gin[m][h] + bnn[h] + r * (ghn[m][h] + bhn[h]));
            float hn = (1.0f - z) * nv + z * hflt[m][h];
            hflt[m][h] = hn;
            hbuf[m][h] = f2bf(hn);
        }
        __syncthreads();
    }

    // ---- write final hidden state [16 x 128] ----
    for (int e = tid; e < 2048; e += 256)
        hout[e] = hflt[e >> 7][e & 127];
}

// ---------------------------------------------------------------------------
// Kernel 2: out[b,c,t,f] = sum_h Wp[c,h]*(hTt[b*256+t,h]+hTf[b*256+t,h]) + bp[c]
// broadcast along f. One block per (b,t); the 134 MB store dominates.
// ---------------------------------------------------------------------------
__global__ __launch_bounds__(256)
void proj_kernel(const float* __restrict__ hTt, const float* __restrict__ hTf,
                 const float* __restrict__ Wp, const float* __restrict__ bp,
                 float* __restrict__ out)
{
    __shared__ float svec[128];
    __shared__ float cval[64];
    const int s   = blockIdx.x;              // b*256 + t
    const int tid = threadIdx.x;

    if (tid < 128) svec[tid] = hTt[(size_t)s * 128 + tid] + hTf[(size_t)s * 128 + tid];
    __syncthreads();

    if (tid < 64) {
        float acc = bp[tid];
        const float* w = Wp + (size_t)tid * 128;
#pragma unroll 8
        for (int h = 0; h < 128; ++h) acc += w[h] * svec[h];
        cval[tid] = acc;
    }
    __syncthreads();

    const int bb = s >> 8, t = s & 255;
    float* ob = out + ((size_t)(bb * 64) * 256 + t) * 256;   // c=0, f=0
    for (int c = 0; c < 64; ++c)
        ob[(size_t)c * 65536 + tid] = cval[c];               // f = tid, coalesced
}

// ---------------------------------------------------------------------------
extern "C" void kernel_launch(void* const* d_in, const int* in_sizes, int n_in,
                              void* d_out, int out_size, void* d_ws, size_t ws_size,
                              hipStream_t stream) {
    const float* x     = (const float*)d_in[0];
    const float* Wih_t = (const float*)d_in[1];
    const float* Whh_t = (const float*)d_in[2];
    const float* bih_t = (const float*)d_in[3];
    const float* bhh_t = (const float*)d_in[4];
    const float* Wih_f = (const float*)d_in[5];
    const float* Whh_f = (const float*)d_in[6];
    const float* bih_f = (const float*)d_in[7];
    const float* bhh_f = (const float*)d_in[8];
    const float* Wp    = (const float*)d_in[9];
    const float* bp    = (const float*)d_in[10];
    float* out = (float*)d_out;

    unsigned short* wbf = (unsigned short*)d_ws;
    unsigned short* wih_t_bf = wbf;                 // 384*64
    unsigned short* whh_t_bf = wbf + 24576;         // 384*128
    unsigned short* wih_f_bf = wbf + 73728;         // 384*64
    unsigned short* whh_f_bf = wbf + 98304;         // 384*128
    float* hTt = (float*)(wbf + 147456);            // 2048*128 f32
    float* hTf = hTt + 2048 * 128;                  // 2048*128 f32

    prep_weights<<<576, 256, 0, stream>>>(Wih_t, Whh_t, Wih_f, Whh_f, wbf);
    gru_kernel<<<256, 256, 0, stream>>>(x,
                                        wih_t_bf, whh_t_bf, bih_t, bhh_t,
                                        wih_f_bf, whh_f_bf, bih_f, bhh_f,
                                        hTt, hTf);
    proj_kernel<<<2048, 256, 0, stream>>>(hTt, hTf, Wp, bp, out);
}
